// PLEASING_32049045963203
// MI455X (gfx1250) — matrix operand order repaired and checked by reading, not compiled, not run on hardware
//
#include <hip/hip_runtime.h>
#include <hip/hip_bf16.h>
#include <math.h>

#define H     200
#define HP    224          // H padded to 7 * 32 (WMMA K) ; also multiple of 16
#define NENT  100000
#define NEDGE 1000000
#define BSZ   512
#define TB    256
#define RRSLOPE ((1.0f/8.0f + 1.0f/3.0f) * 0.5f)
#define BN_EPS  1e-5f

typedef __attribute__((ext_vector_type(16))) _Float16 v16h;
typedef __attribute__((ext_vector_type(8)))  _Float16 v8h;
typedef __attribute__((ext_vector_type(8)))  float    v8f;

// -------------------- helpers --------------------

__device__ __forceinline__ void atomAddF(float* p, float v) {
  unsafeAtomicAdd(p, v);   // native global_atomic_add_f32
}

// A fragment, 16x32 f16, from row-major [., pitch] at (m0, k0).
// Layout: lanes 0-15 hold rows M=0..15 with K = k0..k0+15 contiguous;
// lanes 16-31 hold same rows, K = k0+16..k0+31 (matches load_b).
__device__ __forceinline__ v16h load_a(const _Float16* __restrict__ base, int pitch,
                                       int m0, int k0) {
  int lane = threadIdx.x & 31;
  int half = lane >> 4;
  const _Float16* p = base + (size_t)(m0 + (lane & 15)) * pitch + k0 + half * 16;
  union { v16h v; v8h h[2]; } u;
  u.h[0] = *(const v8h*)p;          // K block [k0+16*half, +8)
  u.h[1] = *(const v8h*)(p + 8);    // K block [k0+16*half+8, +8)
  return u.v;
}

// B fragment, 32x16 f16, from Bt stored row-major as [col][k] with pitch.
// Lane c<16: column n0+c, K = k0..k0+15 contiguous; lane c+16: K = k0+16..k0+31.
__device__ __forceinline__ v16h load_b(const _Float16* __restrict__ bt, int pitch,
                                       int n0, int k0) {
  int lane = threadIdx.x & 31;
  const _Float16* p = bt + (size_t)(n0 + (lane & 15)) * pitch + k0 + (lane >> 4) * 16;
  return *(const v16h*)p;           // 32B aligned: pitch=224 halves, k step 32B
}

// C/D layout: lane L: col = n0+(L&15); VGPR r holds row m0 + 8*(L>>4) + r.
__device__ __forceinline__ void store_tile(float* __restrict__ out, size_t pitch,
                                           int m0, int n0, v8f c) {
  int lane = threadIdx.x & 31;
  int col = n0 + (lane & 15);
  int row = m0 + 8 * (lane >> 4);
#pragma unroll
  for (int r = 0; r < 8; ++r) out[(size_t)(row + r) * pitch + col] = c[r];
}

__device__ __forceinline__ void store_tile_rrelu(float* __restrict__ out, size_t pitch,
                                                 int m0, int n0, int ncol, v8f c) {
  int lane = threadIdx.x & 31;
  int col = n0 + (lane & 15);
  int row = m0 + 8 * (lane >> 4);
  if (col < ncol) {
#pragma unroll
    for (int r = 0; r < 8; ++r) {
      float v = c[r];
      v = (v >= 0.0f) ? v : v * RRSLOPE;
      out[(size_t)(row + r) * pitch + col] = v;
    }
  }
}

// -------------------- kernels --------------------

__global__ void k_zero(float* __restrict__ p, long n) {
  long i = (long)blockIdx.x * blockDim.x + threadIdx.x;
  if (i < n) p[i] = 0.0f;
}

// Wt[m][j][k] = W_m[k][j] (f16, zero-padded to HPxHP). m selects which matrix.
__global__ void k_convert_w(const float* __restrict__ w0, const float* __restrict__ w1,
                            const float* __restrict__ w2, const float* __restrict__ w3,
                            _Float16* __restrict__ wt) {
  int idx = blockIdx.x * blockDim.x + threadIdx.x;
  if (idx >= 4 * HP * HP) return;
  int m = idx / (HP * HP);
  int r = idx - m * (HP * HP);
  int j = r / HP, k = r - j * HP;
  const float* w = (m == 0) ? w0 : (m == 1) ? w1 : (m == 2) ? w2 : w3;
  float v = (j < H && k < H) ? w[k * H + j] : 0.0f;
  wt[idx] = (_Float16)v;
}

// gate[n] = sigmoid(theta[n,:] . gw + gb) ; one wave32 per row.
__global__ void k_gate(const float* __restrict__ theta, const float* __restrict__ gw,
                       const float* __restrict__ gb, float* __restrict__ gate) {
  int wave = (blockIdx.x * blockDim.x + threadIdx.x) >> 5;
  int lane = threadIdx.x & 31;
  if (wave >= NENT) return;
  float s = 0.0f;
  for (int k = lane; k < H; k += 32) s += theta[(size_t)wave * H + k] * gw[k];
#pragma unroll
  for (int off = 16; off > 0; off >>= 1) s += __shfl_xor(s, off, 32);
  if (lane == 0) gate[wave] = 1.0f / (1.0f + expf(-(s + gb[0])));
}

__global__ void k_deg(const int* __restrict__ dst, float* __restrict__ deg) {
  long e = (long)blockIdx.x * blockDim.x + threadIdx.x;
  if (e < NEDGE) atomAddF(&deg[dst[e]], 1.0f);
}

// agg[dst] += h[src] + rel_emb[etype] ; 64 threads per edge (50 active x float4).
__global__ void k_scatter(const float* __restrict__ h, const float* __restrict__ rel,
                          const int* __restrict__ src, const int* __restrict__ dst,
                          const int* __restrict__ etype, float* __restrict__ agg) {
  long tid = (long)blockIdx.x * blockDim.x + threadIdx.x;
  int  c = (int)(tid & 63);
  long e = tid >> 6;
  if (e >= NEDGE || c >= (H / 4)) return;
  float4 hv = *(const float4*)(h   + (size_t)src[e]   * H + c * 4);
  float4 rv = *(const float4*)(rel + (size_t)etype[e] * H + c * 4);
  float* out = agg + (size_t)dst[e] * HP + c * 4;
  atomAddF(out + 0, hv.x + rv.x);
  atomAddF(out + 1, hv.y + rv.y);
  atomAddF(out + 2, hv.z + rv.z);
  atomAddF(out + 3, hv.w + rv.w);
}

// a16 = (agg / max(deg,1)) as f16 padded ; h16 = h as f16 padded.
__global__ void k_norm_cast(const float* __restrict__ agg, const float* __restrict__ deg,
                            const float* __restrict__ h,
                            _Float16* __restrict__ a16, _Float16* __restrict__ h16) {
  long idx = (long)blockIdx.x * blockDim.x + threadIdx.x;
  if (idx >= (long)NENT * HP) return;
  int n = (int)(idx / HP), k = (int)(idx - (long)n * HP);
  if (k < H) {
    float d = deg[n]; d = (d > 1.0f) ? d : 1.0f;
    a16[idx] = (_Float16)(agg[idx] / d);
    h16[idx] = (_Float16)h[(size_t)n * H + k];
  } else {
    a16[idx] = (_Float16)0.0f;
    h16[idx] = (_Float16)0.0f;
  }
}

// out = rrelu(a16 @ Wn^T-stored + h16 @ Wl^T-stored), one 16x16 tile per wave.
__global__ void k_gemm_layer(const _Float16* __restrict__ a16, const _Float16* __restrict__ h16,
                             const _Float16* __restrict__ wtn, const _Float16* __restrict__ wtl,
                             float* __restrict__ out) {
  const int NTJ = (H + 15) / 16;               // 13 column tiles
  int wave = (blockIdx.x * blockDim.x + threadIdx.x) >> 5;
  if (wave >= (NENT / 16) * NTJ) return;        // wave-uniform exit; EXEC stays full
  int tm = wave / NTJ, tn = wave - tm * NTJ;
  int m0 = tm * 16, n0 = tn * 16;
  v8f acc = {};
#pragma unroll
  for (int k0 = 0; k0 < HP; k0 += 32) {
    v16h a = load_a(a16, HP, m0, k0);
    v16h b = load_b(wtn, HP, n0, k0);
    acc = __builtin_amdgcn_wmma_f32_16x16x32_f16(false, a, false, b, (short)0, acc, false, false);
    a = load_a(h16, HP, m0, k0);
    b = load_b(wtl, HP, n0, k0);
    acc = __builtin_amdgcn_wmma_f32_16x16x32_f16(false, a, false, b, (short)0, acc, false, false);
  }
  store_tile_rrelu(out, H, m0, n0, H, acc);
}

// h_final = gate*h2 + (1-gate)*ent (f32, in place into h2) + padded f16 copy.
__global__ void k_final(const float* __restrict__ gate, const float* __restrict__ ent,
                        float* __restrict__ h2, _Float16* __restrict__ hf16) {
  long idx = (long)blockIdx.x * blockDim.x + threadIdx.x;
  if (idx >= (long)NENT * HP) return;
  int n = (int)(idx / HP), k = (int)(idx - (long)n * HP);
  if (k < H) {
    float g = gate[n];
    size_t i = (size_t)n * H + k;
    float f = g * h2[i] + (1.0f - g) * ent[i];
    h2[i] = f;
    hf16[idx] = (_Float16)f;
  } else {
    hf16[idx] = (_Float16)0.0f;
  }
}

// BatchNorm1d(3) batch statistics over (B,H) per channel. One block.
__global__ void k_bnstats(const float* __restrict__ hf, const float* __restrict__ rel,
                          const float* __restrict__ freq, const int* __restrict__ s_idx,
                          const int* __restrict__ r_idx, float* __restrict__ stat) {
  __shared__ float sh[6][TB];
  int t = threadIdx.x;
  float s0 = 0, s1 = 0, s2 = 0, q0 = 0, q1 = 0, q2 = 0;
  for (int i = t; i < BSZ * H; i += TB) {
    int b = i / H, h = i - b * H;
    float e = hf[(size_t)s_idx[b] * H + h];
    float r = rel[(size_t)r_idx[b] * H + h];
    float f = freq[i];
    s0 += e; q0 += e * e;
    s1 += r; q1 += r * r;
    s2 += f; q2 += f * f;
  }
  sh[0][t] = s0; sh[1][t] = s1; sh[2][t] = s2;
  sh[3][t] = q0; sh[4][t] = q1; sh[5][t] = q2;
  __syncthreads();
  for (int s = TB / 2; s > 0; s >>= 1) {
    if (t < s) {
#pragma unroll
      for (int c = 0; c < 6; ++c) sh[c][t] += sh[c][t + s];
    }
    __syncthreads();
  }
  if (t < 3) {
    float inv = 1.0f / (float)(BSZ * H);
    float mean = sh[t][0] * inv;
    float var  = sh[t + 3][0] * inv - mean * mean;
    stat[t]     = mean;
    stat[3 + t] = rsqrtf(var + BN_EPS);
  }
}

// q[b,h] = sum_c relu(BN(x_c)) * conv_w[c] + conv_b, emitted as padded f16.
__global__ void k_q(const float* __restrict__ hf, const float* __restrict__ rel,
                    const float* __restrict__ freq, const int* __restrict__ s_idx,
                    const int* __restrict__ r_idx, const float* __restrict__ stat,
                    const float* __restrict__ gamma, const float* __restrict__ beta,
                    const float* __restrict__ convw, const float* __restrict__ convb,
                    _Float16* __restrict__ q16) {
  int idx = blockIdx.x * blockDim.x + threadIdx.x;
  if (idx >= BSZ * HP) return;
  int b = idx / HP, k = idx - b * HP;
  if (k >= H) { q16[idx] = (_Float16)0.0f; return; }
  float x[3];
  x[0] = hf[(size_t)s_idx[b] * H + k];
  x[1] = rel[(size_t)r_idx[b] * H + k];
  x[2] = freq[b * H + k];
  float q = convb[0];
#pragma unroll
  for (int c = 0; c < 3; ++c) {
    float v = (x[c] - stat[c]) * stat[3 + c] * gamma[c] + beta[c];
    v = (v > 0.0f) ? v : 0.0f;
    q += v * convw[c];
  }
  q16[idx] = (_Float16)q;
}

// scores[512,100000] = q16 @ hf16^T. Wave handles 4 M-tiles x 1 N-tile,
// reusing the streamed B fragment (hf16 is already [col][k] row-major = Bt).
__global__ void k_gemm_scores(const _Float16* __restrict__ q16,
                              const _Float16* __restrict__ hf16,
                              float* __restrict__ out) {
  const int NTN = NENT / 16;                       // 6250
  int wave = (blockIdx.x * blockDim.x + threadIdx.x) >> 5;
  if (wave >= (BSZ / 64) * NTN) return;            // 8 * 6250 waves
  int tmb = wave / NTN, tn = wave - tmb * NTN;
  int n0 = tn * 16;
  int m0 = tmb * 64;
  v8f a0 = {}, a1 = {}, a2 = {}, a3 = {};
#pragma unroll
  for (int k0 = 0; k0 < HP; k0 += 32) {
    __builtin_prefetch(hf16 + (size_t)(n0 + 16) * HP + k0, 0, 1);
    v16h b  = load_b(hf16, HP, n0, k0);
    v16h fa = load_a(q16, HP, m0 +  0, k0);
    a0 = __builtin_amdgcn_wmma_f32_16x16x32_f16(false, fa, false, b, (short)0, a0, false, false);
    fa = load_a(q16, HP, m0 + 16, k0);
    a1 = __builtin_amdgcn_wmma_f32_16x16x32_f16(false, fa, false, b, (short)0, a1, false, false);
    fa = load_a(q16, HP, m0 + 32, k0);
    a2 = __builtin_amdgcn_wmma_f32_16x16x32_f16(false, fa, false, b, (short)0, a2, false, false);
    fa = load_a(q16, HP, m0 + 48, k0);
    a3 = __builtin_amdgcn_wmma_f32_16x16x32_f16(false, fa, false, b, (short)0, a3, false, false);
  }
  store_tile(out, NENT, m0 +  0, n0, a0);
  store_tile(out, NENT, m0 + 16, n0, a1);
  store_tile(out, NENT, m0 + 32, n0, a2);
  store_tile(out, NENT, m0 + 48, n0, a3);
}

// -------------------- launcher --------------------

extern "C" void kernel_launch(void* const* d_in, const int* in_sizes, int n_in,
                              void* d_out, int out_size, void* d_ws, size_t ws_size,
                              hipStream_t stream) {
  const float* ent   = (const float*)d_in[0];
  const float* rel   = (const float*)d_in[1];
  const float* Wn1   = (const float*)d_in[2];
  const float* Wl1   = (const float*)d_in[3];
  const float* Wn2   = (const float*)d_in[4];
  const float* Wl2   = (const float*)d_in[5];
  const float* gth   = (const float*)d_in[6];
  const float* gw    = (const float*)d_in[7];
  const float* gb    = (const float*)d_in[8];
  const float* gamma = (const float*)d_in[9];
  const float* beta  = (const float*)d_in[10];
  const float* convw = (const float*)d_in[11];
  const float* convb = (const float*)d_in[12];
  const float* freq  = (const float*)d_in[13];
  const int*   src   = (const int*)d_in[14];
  const int*   dst   = (const int*)d_in[15];
  const int*   etype = (const int*)d_in[16];
  const int*   s_idx = (const int*)d_in[17];
  const int*   r_idx = (const int*)d_in[18];
  float* scores = (float*)d_out;

  char* base = (char*)d_ws;
  size_t off = 0;
  auto take = [&](size_t bytes) -> char* {
    char* p = base + off;
    off = (off + bytes + 255) & ~(size_t)255;
    return p;
  };
  _Float16* wt   = (_Float16*)take(4ull * HP * HP * sizeof(_Float16));
  float*    deg  = (float*)   take((size_t)NENT * sizeof(float));
  float*    gate = (float*)   take((size_t)NENT * sizeof(float));
  float*    stat = (float*)   take(8 * sizeof(float));
  _Float16* q16  = (_Float16*)take((size_t)BSZ * HP * sizeof(_Float16));
  float*    agg  = (float*)   take((size_t)NENT * HP * sizeof(float));
  _Float16* a16  = (_Float16*)take((size_t)NENT * HP * sizeof(_Float16));
  _Float16* x16  = (_Float16*)take((size_t)NENT * HP * sizeof(_Float16)); // h f16 / h_final f16
  float*    h1   = (float*)   take((size_t)NENT * H * sizeof(float));
  float*    h2   = (float*)   take((size_t)NENT * H * sizeof(float));

  const long npad = (long)NENT * HP;
  const int  ntl  = (NENT / 16) * ((H + 15) / 16);   // 81250 layer-GEMM tiles
  const int  nws  = (BSZ / 64) * (NENT / 16);        // 50000 scores-GEMM waves

  // init (ws is poisoned; re-zero accumulators every call)
  k_zero<<<(int)((npad + TB - 1) / TB), TB, 0, stream>>>(agg, npad);
  k_zero<<<(NENT + TB - 1) / TB, TB, 0, stream>>>(deg, NENT);
  k_convert_w<<<(4 * HP * HP + TB - 1) / TB, TB, 0, stream>>>(Wn1, Wl1, Wn2, Wl2, wt);
  k_gate<<<(NENT * 32 + TB - 1) / TB, TB, 0, stream>>>(gth, gw, gb, gate);
  k_deg<<<(NEDGE + TB - 1) / TB, TB, 0, stream>>>(dst, deg);

  // layer 1
  k_scatter<<<(int)(((long)NEDGE * 64) / TB), TB, 0, stream>>>(ent, rel, src, dst, etype, agg);
  k_norm_cast<<<(int)((npad + TB - 1) / TB), TB, 0, stream>>>(agg, deg, ent, a16, x16);
  k_gemm_layer<<<(ntl * 32 + TB - 1) / TB, TB, 0, stream>>>(a16, x16, wt, wt + (size_t)HP * HP, h1);

  // layer 2
  k_zero<<<(int)((npad + TB - 1) / TB), TB, 0, stream>>>(agg, npad);
  k_scatter<<<(int)(((long)NEDGE * 64) / TB), TB, 0, stream>>>(h1, rel, src, dst, etype, agg);
  k_norm_cast<<<(int)((npad + TB - 1) / TB), TB, 0, stream>>>(agg, deg, h1, a16, x16);
  k_gemm_layer<<<(ntl * 32 + TB - 1) / TB, TB, 0, stream>>>(a16, x16, wt + 2ull * HP * HP, wt + 3ull * HP * HP, h2);

  // gating blend (h_final f32 in h2, f16 in x16)
  k_final<<<(int)((npad + TB - 1) / TB), TB, 0, stream>>>(gate, ent, h2, x16);

  // ConvE head
  k_bnstats<<<1, TB, 0, stream>>>(h2, rel, freq, s_idx, r_idx, stat);
  k_q<<<(BSZ * HP + TB - 1) / TB, TB, 0, stream>>>(h2, rel, freq, s_idx, r_idx, stat,
                                                   gamma, beta, convw, convb, q16);

  // scores = q @ h_final^T
  k_gemm_scores<<<(nws * 32 + TB - 1) / TB, TB, 0, stream>>>(q16, x16, scores);
}
